// TinyTransformer_31851477467875
// MI455X (gfx1250) — compile-verified, matrix operand order checked by reading, not run on hardware
//
#include <hip/hip_runtime.h>
#include <stdint.h>

#define S_LEN 2048
#define DM    2048
#define NH    16
#define HD    128
#define DFF   8192
#define NV    32000

typedef __bf16 bf16;
typedef bf16  v8bf  __attribute__((ext_vector_type(8)));
typedef bf16  v16bf __attribute__((ext_vector_type(16)));
typedef float v8f   __attribute__((ext_vector_type(8)));
typedef int   v4i   __attribute__((vector_size(16)));   // matches builtin proto

// ---------------------------------------------------------------------------
// gfx1250 async global->LDS copy (ASYNCcnt path), with synchronous fallback.
// Builtin signature (from compiler diagnostic): (int4 as1*, int4 as3*, Ii, Ii)
// ---------------------------------------------------------------------------
#if defined(__has_builtin)
#if __has_builtin(__builtin_amdgcn_global_load_async_to_lds_b128)
#define USE_ASYNC_LDS 1
#endif
#endif
#ifndef USE_ASYNC_LDS
#define USE_ASYNC_LDS 0
#endif

static __device__ inline void async_copy16(const bf16* __restrict__ g, bf16* l) {
#if USE_ASYNC_LDS
  // generic->AS cast via uintptr_t: global generic addr == AS1 addr; LDS
  // generic addr keeps the LDS offset in addr[31:0] (flat aperture rule).
  __builtin_amdgcn_global_load_async_to_lds_b128(
      (__attribute__((address_space(1))) v4i*)(uintptr_t)g,
      (__attribute__((address_space(3))) v4i*)(uintptr_t)l,
      /*offset=*/0, /*cpol=*/0);
#else
  *(v8bf*)l = *(const v8bf*)g;
#endif
}

static __device__ inline void async_wait() {
#if USE_ASYNC_LDS
#if __has_builtin(__builtin_amdgcn_s_wait_asynccnt)
  __builtin_amdgcn_s_wait_asynccnt(0);
#else
  asm volatile("s_wait_asynccnt 0x0" ::: "memory");
#endif
#endif
}

// ---------------------------------------------------------------------------
// WMMA helper:  D(16x16,f32) = A(16x32,bf16) * B(32x16,bf16) + C
// ---------------------------------------------------------------------------
static __device__ inline v8f wmma_bf16(v16bf a, v16bf b, v8f c) {
  return __builtin_amdgcn_wmma_f32_16x16x32_bf16(
      false, a, false, b, (short)0, c, false, false);
}

// Operand fragment for 16x16x32 bf16 WMMA: lanes 0-15 hold K={0..7,16..23},
// lanes 16-31 hold K={8..15,24..31}: two contiguous 16B chunks per lane.
static __device__ inline v16bf load_frag(const bf16* p, int hi) {
  v8bf lo = *(const v8bf*)(p + hi * 8);
  v8bf hh = *(const v8bf*)(p + 16 + hi * 8);
  return __builtin_shufflevector(lo, hh, 0,1,2,3,4,5,6,7,8,9,10,11,12,13,14,15);
}

// ---------------------------------------------------------------------------
// One-time weight convert+transpose: W f32 [K][N]  ->  Wt bf16 [N][K]
// ---------------------------------------------------------------------------
__global__ __launch_bounds__(256) void transpose_cvt(
    const float* __restrict__ W, bf16* __restrict__ Wt, int K, int N) {
  __shared__ bf16 tile[32][33];
  const int n0 = blockIdx.x * 32, k0 = blockIdx.y * 32;
#pragma unroll
  for (int i = 0; i < 4; ++i) {
    const int kk = threadIdx.y + i * 8;
    tile[kk][threadIdx.x] = (bf16)W[(size_t)(k0 + kk) * N + n0 + threadIdx.x];
  }
  __syncthreads();
#pragma unroll
  for (int i = 0; i < 4; ++i) {
    const int nn = threadIdx.y + i * 8;
    Wt[(size_t)(n0 + nn) * K + k0 + threadIdx.x] = tile[threadIdx.x][nn];
  }
}

// ---------------------------------------------------------------------------
// x[s][:] = token_emb[tokens[s]][:] + pos_emb[s][:]
// ---------------------------------------------------------------------------
__global__ __launch_bounds__(256) void embed_kernel(
    const long long* __restrict__ toks, const float* __restrict__ temb,
    const float* __restrict__ pemb, float* __restrict__ x) {
  const int srow = blockIdx.x;
  const long long tok = toks[srow];
  const float* te = temb + (size_t)tok * DM;
  const float* pe = pemb + (size_t)srow * DM;
  float* xo = x + (size_t)srow * DM;
#pragma unroll
  for (int i = 0; i < DM / 256; ++i) {
    const int d = i * 256 + threadIdx.x;
    xo[d] = te[d] + pe[d];
  }
}

// ---------------------------------------------------------------------------
// LayerNorm over last dim (2048), block per row, bf16 output
// ---------------------------------------------------------------------------
__global__ __launch_bounds__(256) void ln_kernel(
    const float* __restrict__ x, const float* __restrict__ g,
    const float* __restrict__ bta, bf16* __restrict__ out) {
  __shared__ float red[256];
  const int row = blockIdx.x, t = threadIdx.x;
  const float* xr = x + (size_t)row * DM;
  float v[DM / 256];
  float s = 0.f;
#pragma unroll
  for (int i = 0; i < DM / 256; ++i) { v[i] = xr[i * 256 + t]; s += v[i]; }
  red[t] = s; __syncthreads();
  for (int o = 128; o > 0; o >>= 1) { if (t < o) red[t] += red[t + o]; __syncthreads(); }
  const float mean = red[0] * (1.f / DM);
  __syncthreads();
  s = 0.f;
#pragma unroll
  for (int i = 0; i < DM / 256; ++i) { const float d = v[i] - mean; s += d * d; }
  red[t] = s; __syncthreads();
  for (int o = 128; o > 0; o >>= 1) { if (t < o) red[t] += red[t + o]; __syncthreads(); }
  const float rstd = rsqrtf(red[0] * (1.f / DM) + 1e-5f);
  bf16* orow = out + (size_t)row * DM;
#pragma unroll
  for (int i = 0; i < DM / 256; ++i) {
    const int d = i * 256 + t;
    orow[d] = (bf16)((v[i] - mean) * rstd * g[d] + bta[d]);
  }
}

// ---------------------------------------------------------------------------
// NT GEMM: D[M][N] = A[M][K](bf16) * Bt[N][K](bf16)^T  (+C, +relu variants)
// 128x128 block tile, 8 waves each 64x32.  K staged 64-wide via async LDS
// DMA into double-buffered tiles: DMA of tile i+1 overlaps WMMA on tile i.
// ---------------------------------------------------------------------------
template <int ADD_RES, int RELU, int OUT_BF16>
__global__ __launch_bounds__(256) void gemm_nt(
    const bf16* __restrict__ A, const bf16* __restrict__ Bt,
    const float* __restrict__ Cres, float* __restrict__ Df,
    bf16* __restrict__ Db, int M, int N, int K) {
  __shared__ __align__(16) bf16 As[2][128][64];   // 64 KB total
  __shared__ __align__(16) bf16 Bs[2][128][64];
  const int t = threadIdx.x, wid = t >> 5, lane = t & 31;
  const int hi = lane >> 4, ln = lane & 15;
  const int m0 = blockIdx.y * 128, n0 = blockIdx.x * 128;
  const int wr = (wid >> 2) * 64, wc = (wid & 3) * 32;

  auto stage = [&](int buf, int kc) {
#pragma unroll
    for (int i = 0; i < 4; ++i) {                 // 128x64 tiles: 1024 16B
      const int c = t + i * 256;                  // chunks per array
      const int r = c >> 3, qq = c & 7;
      async_copy16(A + (size_t)(m0 + r) * K + kc + qq * 8, &As[buf][r][qq * 8]);
      async_copy16(Bt + (size_t)(n0 + r) * K + kc + qq * 8, &Bs[buf][r][qq * 8]);
    }
  };

  v8f acc[4][2];
  const v8f vz = {0.f, 0.f, 0.f, 0.f, 0.f, 0.f, 0.f, 0.f};
#pragma unroll
  for (int a = 0; a < 4; ++a)
#pragma unroll
    for (int b = 0; b < 2; ++b) acc[a][b] = vz;

  stage(0, 0);
  int buf = 0;
  for (int kc = 0; kc < K; kc += 64) {
    async_wait();                                 // tile kc resident (own DMAs)
    __syncthreads();                              // ... for ALL waves
    if (kc + 64 < K) stage(buf ^ 1, kc + 64);     // overlap DMA with WMMA
#pragma unroll
    for (int kk = 0; kk < 2; ++kk) {              // two 32-wide K steps
      v16bf bfrag[2];
#pragma unroll
      for (int sn = 0; sn < 2; ++sn)
        bfrag[sn] = load_frag(&Bs[buf][wc + sn * 16 + ln][kk * 32], hi);
#pragma unroll
      for (int sm = 0; sm < 4; ++sm) {
        const v16bf afrag = load_frag(&As[buf][wr + sm * 16 + ln][kk * 32], hi);
        acc[sm][0] = wmma_bf16(afrag, bfrag[0], acc[sm][0]);
        acc[sm][1] = wmma_bf16(afrag, bfrag[1], acc[sm][1]);
      }
    }
    buf ^= 1;
  }

#pragma unroll
  for (int sm = 0; sm < 4; ++sm)
#pragma unroll
    for (int sn = 0; sn < 2; ++sn)
#pragma unroll
      for (int j = 0; j < 8; ++j) {               // C layout: row = j + 8*hi
        const size_t r = m0 + wr + sm * 16 + j + hi * 8;
        const size_t cix = n0 + wc + sn * 16 + ln;
        float val = acc[sm][sn][j];
        if (RELU) val = fmaxf(val, 0.f);
        if (ADD_RES) val += Cres[r * N + cix];
        if (OUT_BF16) {
          Db[r * N + cix] = (bf16)val;
        } else if (ADD_RES) {
          Df[r * N + cix] = val;
        } else {                                  // logits: streamed, bypass L2
          __builtin_nontemporal_store(val, &Df[r * N + cix]);
        }
      }
}

// ---------------------------------------------------------------------------
// Flash attention: block = (head, 128 queries); wave = 16 queries.
// Double-buffered 32-key K/V^T staging overlapping the WMMA/softmax work.
// ---------------------------------------------------------------------------
__global__ __launch_bounds__(256) void attn_kernel(
    const bf16* __restrict__ q, const bf16* __restrict__ k,
    const bf16* __restrict__ v, bf16* __restrict__ ao) {
  __shared__ __align__(16) bf16 Ks[2][32][128];   // [key][hd]
  __shared__ __align__(16) bf16 Vt[2][128][32];   // [hd][key]
  __shared__ __align__(16) bf16 Ps[8][16][32];    // per-wave P tile [m][key]
  const int t = threadIdx.x, wid = t >> 5, lane = t & 31;
  const int hi = lane >> 4, ln = lane & 15;
  const int head = blockIdx.x / (S_LEN / 128);
  const int qbase = (blockIdx.x % (S_LEN / 128)) * 128 + wid * 16;
  const size_t hoff = (size_t)head * HD;

  auto stage = [&](int buf, int t0) {
#pragma unroll
    for (int i = 0; i < 2; ++i) {                 // 32x128 tiles: 512 16B chunks
      const int c = t + i * 256;
      const int r = c >> 4, c8 = c & 15;
      async_copy16(k + (size_t)(t0 + r) * DM + hoff + c8 * 8, &Ks[buf][r][c8 * 8]);
      const v8bf vv = *(const v8bf*)(v + (size_t)(t0 + r) * DM + hoff + c8 * 8);
#pragma unroll
      for (int e = 0; e < 8; ++e) Vt[buf][c8 * 8 + e][r] = vv[e];  // transpose V
    }
  };

  v16bf qf[4];                                    // resident Q, hd = 4 x 32
#pragma unroll
  for (int kc = 0; kc < 4; ++kc)
    qf[kc] = load_frag(q + (size_t)(qbase + ln) * DM + hoff + kc * 32, hi);

  v8f acc[8];                                     // 16 x 128 f32 output
  const v8f vz = {0.f, 0.f, 0.f, 0.f, 0.f, 0.f, 0.f, 0.f};
#pragma unroll
  for (int c = 0; c < 8; ++c) acc[c] = vz;
  float rm[8], rl[8];
#pragma unroll
  for (int j = 0; j < 8; ++j) { rm[j] = -1e30f; rl[j] = 0.f; }
  const float sc = 0.08838834764831845f;          // 1/sqrt(128)

  stage(0, 0);
  int buf = 0;
  for (int t0 = 0; t0 < S_LEN; t0 += 32) {
    async_wait();
    __syncthreads();
    if (t0 + 32 < S_LEN) stage(buf ^ 1, t0 + 32); // overlap next K/V DMA

    v8f s0 = vz, s1 = vz;                         // scores vs keys t0..t0+31
#pragma unroll
    for (int kc = 0; kc < 4; ++kc) {
      const v16bf b0 = load_frag(&Ks[buf][ln][kc * 32], hi);
      const v16bf b1 = load_frag(&Ks[buf][16 + ln][kc * 32], hi);
      s0 = wmma_bf16(qf[kc], b0, s0);
      s1 = wmma_bf16(qf[kc], b1, s1);
    }

    float al[8];
#pragma unroll
    for (int j = 0; j < 8; ++j) {                 // row = j + 8*hi
      const float a = s0[j] * sc, b = s1[j] * sc;
      float mx = fmaxf(a, b);
#pragma unroll
      for (int m = 1; m < 16; m <<= 1) mx = fmaxf(mx, __shfl_xor(mx, m, 32));
      const float nm = fmaxf(rm[j], mx);
      const float alpha = __expf(rm[j] - nm);
      const float p0 = __expf(a - nm), p1 = __expf(b - nm);
      float ps = p0 + p1;
#pragma unroll
      for (int m = 1; m < 16; m <<= 1) ps += __shfl_xor(ps, m, 32);
      rl[j] = rl[j] * alpha + ps;
      rm[j] = nm;
      al[j] = alpha;
      Ps[wid][j + hi * 8][ln]      = (bf16)p0;    // C layout -> A layout via LDS
      Ps[wid][j + hi * 8][16 + ln] = (bf16)p1;
    }
#pragma unroll
    for (int c = 0; c < 8; ++c)
#pragma unroll
      for (int j = 0; j < 8; ++j) acc[c][j] *= al[j];

    const v16bf pa = load_frag(&Ps[wid][ln][0], hi);
#pragma unroll
    for (int c = 0; c < 8; ++c) {
      const v16bf bv = load_frag(&Vt[buf][c * 16 + ln][0], hi);
      acc[c] = wmma_bf16(pa, bv, acc[c]);
    }
    buf ^= 1;
  }

#pragma unroll
  for (int j = 0; j < 8; ++j) rl[j] = 1.f / rl[j];
#pragma unroll
  for (int c = 0; c < 8; ++c)
#pragma unroll
    for (int j = 0; j < 8; ++j) {
      const size_t r = qbase + j + hi * 8;
      ao[r * DM + hoff + c * 16 + ln] = (bf16)(acc[c][j] * rl[j]);
    }
}

// ---------------------------------------------------------------------------
extern "C" void kernel_launch(void* const* d_in, const int* in_sizes, int n_in,
                              void* d_out, int out_size, void* d_ws, size_t ws_size,
                              hipStream_t stream) {
  const long long* toks = (const long long*)d_in[0];
  const float* temb = (const float*)d_in[1];
  const float* pemb = (const float*)d_in[2];
  const float* wq   = (const float*)d_in[3];
  const float* wk   = (const float*)d_in[4];
  const float* wv   = (const float*)d_in[5];
  const float* wo   = (const float*)d_in[6];
  const float* wfi  = (const float*)d_in[7];
  const float* wfo  = (const float*)d_in[8];
  const float* ln1g = (const float*)d_in[9];
  const float* ln1b = (const float*)d_in[10];
  const float* ln2g = (const float*)d_in[11];
  const float* ln2b = (const float*)d_in[12];
  const float* lnfg = (const float*)d_in[13];
  const float* lnfb = (const float*)d_in[14];
  const float* wout = (const float*)d_in[15];

  char* ws = (char*)d_ws;
  size_t off = 0;
  auto alloc = [&](size_t bytes) -> char* {
    char* p = ws + off;
    off += (bytes + 255) & ~(size_t)255;
    return p;
  };
  const size_t SD4 = (size_t)S_LEN * DM * 4, SD2 = (size_t)S_LEN * DM * 2;
  float* x    = (float*)alloc(SD4);
  float* x2   = (float*)alloc(SD4);
  bf16* h     = (bf16*)alloc(SD2);
  bf16* qb    = (bf16*)alloc(SD2);
  bf16* kb    = (bf16*)alloc(SD2);
  bf16* vb    = (bf16*)alloc(SD2);
  bf16* aob   = (bf16*)alloc(SD2);
  bf16* h2    = (bf16*)alloc(SD2);
  bf16* hf    = (bf16*)alloc(SD2);
  bf16* mid   = (bf16*)alloc((size_t)S_LEN * DFF * 2);
  bf16* wqT   = (bf16*)alloc((size_t)DM * DM * 2);
  bf16* wkT   = (bf16*)alloc((size_t)DM * DM * 2);
  bf16* wvT   = (bf16*)alloc((size_t)DM * DM * 2);
  bf16* woT   = (bf16*)alloc((size_t)DM * DM * 2);
  bf16* wfiT  = (bf16*)alloc((size_t)DFF * DM * 2);
  bf16* wfoT  = (bf16*)alloc((size_t)DM * DFF * 2);
  bf16* woutT = (bf16*)alloc((size_t)NV * DM * 2);

  const dim3 tb(32, 8);
  transpose_cvt<<<dim3(DM / 32, DM / 32), tb, 0, stream>>>(wq, wqT, DM, DM);
  transpose_cvt<<<dim3(DM / 32, DM / 32), tb, 0, stream>>>(wk, wkT, DM, DM);
  transpose_cvt<<<dim3(DM / 32, DM / 32), tb, 0, stream>>>(wv, wvT, DM, DM);
  transpose_cvt<<<dim3(DM / 32, DM / 32), tb, 0, stream>>>(wo, woT, DM, DM);
  transpose_cvt<<<dim3(DFF / 32, DM / 32), tb, 0, stream>>>(wfi, wfiT, DM, DFF);
  transpose_cvt<<<dim3(DM / 32, DFF / 32), tb, 0, stream>>>(wfo, wfoT, DFF, DM);
  transpose_cvt<<<dim3(NV / 32, DM / 32), tb, 0, stream>>>(wout, woutT, DM, NV);

  embed_kernel<<<S_LEN, 256, 0, stream>>>(toks, temb, pemb, x);
  ln_kernel<<<S_LEN, 256, 0, stream>>>(x, ln1g, ln1b, h);

  gemm_nt<0, 0, 1><<<dim3(DM / 128, S_LEN / 128), 256, 0, stream>>>(
      h, wqT, nullptr, nullptr, qb, S_LEN, DM, DM);
  gemm_nt<0, 0, 1><<<dim3(DM / 128, S_LEN / 128), 256, 0, stream>>>(
      h, wkT, nullptr, nullptr, kb, S_LEN, DM, DM);
  gemm_nt<0, 0, 1><<<dim3(DM / 128, S_LEN / 128), 256, 0, stream>>>(
      h, wvT, nullptr, nullptr, vb, S_LEN, DM, DM);

  attn_kernel<<<NH * (S_LEN / 128), 256, 0, stream>>>(qb, kb, vb, aob);

  gemm_nt<1, 0, 0><<<dim3(DM / 128, S_LEN / 128), 256, 0, stream>>>(
      aob, woT, x, x2, nullptr, S_LEN, DM, DM);

  ln_kernel<<<S_LEN, 256, 0, stream>>>(x2, ln2g, ln2b, h2);
  gemm_nt<0, 1, 1><<<dim3(DFF / 128, S_LEN / 128), 256, 0, stream>>>(
      h2, wfiT, nullptr, nullptr, mid, S_LEN, DFF, DM);
  gemm_nt<1, 0, 0><<<dim3(DM / 128, S_LEN / 128), 256, 0, stream>>>(
      mid, wfoT, x2, x, nullptr, S_LEN, DM, DFF);

  ln_kernel<<<S_LEN, 256, 0, stream>>>(x, lnfg, lnfb, hf);
  gemm_nt<0, 0, 0><<<dim3(NV / 128, S_LEN / 128), 256, 0, stream>>>(
      hf, woutT, nullptr, (float*)d_out, nullptr, S_LEN, NV, DM);
}
